// UnimodalDebiasModule_1640677507603
// MI455X (gfx1250) — compile-verified, matrix operand order checked by reading, not run on hardware
//
#include <hip/hip_runtime.h>
#include <hip/hip_bf16.h>

// ---------------- model constants ----------------
#define BSZ      8
#define SEQL     1024
#define DM       256
#define DI       512          // D_INNER
#define NH       8            // NHEADS
#define HD       64           // HEADDIM
#define DSTATE   64
#define CONVD    640          // D_INNER + 2*D_STATE
#define MROWS    (BSZ*SEQL)   // 8192
#define ZX_REAL  1160         // 2*DI + 2*DSTATE + NH
#define ZXW      1216         // padded to /64 for 16x64 C-strips
#define EPSF     1e-6f

typedef __bf16 bf16;
typedef bf16  v16bf __attribute__((ext_vector_type(16)));
typedef bf16  v8bf  __attribute__((ext_vector_type(8)));
typedef float v8f   __attribute__((ext_vector_type(8)));

// ---------------- small device helpers ----------------
__device__ __forceinline__ bf16 tobf(float f) {
  unsigned u = __float_as_uint(f);
  unsigned r = u + 0x7FFFu + ((u >> 16) & 1u);      // RNE truncation
  unsigned short s = (unsigned short)(r >> 16);
  return __builtin_bit_cast(bf16, s);
}
__device__ __forceinline__ bf16 bf0() { return __builtin_bit_cast(bf16, (unsigned short)0); }
__device__ __forceinline__ float siluf(float x) { return x / (1.f + expf(-x)); }
__device__ __forceinline__ float apply_act(float v, int act) {
  switch (act) {
    case 1: return v / (1.f + expf(-v));   // silu
    case 2: return tanhf(v);               // tanh
    case 3: return 1.f / (1.f + expf(-v)); // sigmoid
    default: return v;
  }
}
__device__ __forceinline__ v8f zero8() {
  v8f a;
#pragma unroll
  for (int i = 0; i < 8; ++i) a[i] = 0.f;
  return a;
}
// Per-lane fragment: two contiguous 8-elem (16B) chunks of one row.
__device__ __forceinline__ v16bf load_frag(const bf16* row, int off0, int off1) {
  v8bf lo = *(const v8bf*)(row + off0);
  v8bf hi = *(const v8bf*)(row + off1);
  v16bf r;
#pragma unroll
  for (int i = 0; i < 8; ++i) { r[i] = lo[i]; r[8 + i] = hi[i]; }
  return r;
}

// ---------------- WMMA GEMM:  C[M,N] = act(A[M,K] @ W[N,K]^T + bias) ----------------
// A, W bf16 row-major (ld = K). One wave computes a 16x64 C strip (4 accumulators),
// reusing one A fragment across 4 B fragments; 8 waves/block. Requires N % 64 == 0.
// bf16 A/B VGPR layout (ISA 7.12.2): lane l -> row (l&15); elems e=0..7 -> k = 8*(l>>4)+e,
// e=8..15 -> k = 16 + 8*(l>>4) + (e-8)   => two contiguous 16B loads per fragment.
__global__ void wmma_gemm_kernel(const bf16* __restrict__ A, const bf16* __restrict__ W,
                                 const float* __restrict__ bias,
                                 float* __restrict__ Cf, bf16* __restrict__ Cb,
                                 int M, int N, int K, int act) {
  const int wave = threadIdx.x >> 5;
  const int lane = threadIdx.x & 31;
  const int stripsN = N >> 6;
  const int strip = blockIdx.x * 8 + wave;
  if (strip >= (M >> 4) * stripsN) return;
  const int tm = strip / stripsN, tn = strip - tm * stripsN;
  const int khalf = lane >> 4, r15 = lane & 15;
  const bf16* Arow = A + (size_t)(tm * 16 + r15) * K;
  const bf16* Wrow = W + (size_t)(tn * 64 + r15) * K;
  const size_t wstep = (size_t)16 * K;
  v8f acc0 = zero8(), acc1 = zero8(), acc2 = zero8(), acc3 = zero8();
  for (int k0 = 0; k0 < K; k0 += 32) {
    const int off0 = k0 + 8 * khalf;
    const int off1 = k0 + 16 + 8 * khalf;
    // WGP-scope speculative prefetch of next A chunk (global_prefetch_b8, pulls into
    // all cache levels; OOB speculative prefetches are silently dropped per ISA 10.5)
    __builtin_prefetch((const void*)(Arow + k0 + 256), 0, 3);
    v16bf a  = load_frag(Arow, off0, off1);
    v16bf b0 = load_frag(Wrow,             off0, off1);
    v16bf b1 = load_frag(Wrow + wstep,     off0, off1);
    v16bf b2 = load_frag(Wrow + 2 * wstep, off0, off1);
    v16bf b3 = load_frag(Wrow + 3 * wstep, off0, off1);
    acc0 = __builtin_amdgcn_wmma_f32_16x16x32_bf16(false, a, false, b0, (short)0, acc0, false, false);
    acc1 = __builtin_amdgcn_wmma_f32_16x16x32_bf16(false, a, false, b1, (short)0, acc1, false, false);
    acc2 = __builtin_amdgcn_wmma_f32_16x16x32_bf16(false, a, false, b2, (short)0, acc2, false, false);
    acc3 = __builtin_amdgcn_wmma_f32_16x16x32_bf16(false, a, false, b3, (short)0, acc3, false, false);
  }
  v8f accs[4] = {acc0, acc1, acc2, acc3};
#pragma unroll
  for (int s = 0; s < 4; ++s) {
    const int n = tn * 64 + s * 16 + r15;
    const float bv = bias ? bias[n] : 0.f;
#pragma unroll
    for (int r = 0; r < 8; ++r) {
      const int m = tm * 16 + r + 8 * khalf;  // C layout: VGPR r -> M=r (+8 for hi lanes)
      float v = apply_act(accs[s][r] + bv, act);
      const size_t idx = (size_t)m * N + n;
      if (Cf) Cf[idx] = v;
      if (Cb) Cb[idx] = tobf(v);
    }
  }
}

// ---------------- depthwise causal conv(4) + silu over zxbcdt's xBC slice ----------------
__global__ void k_dwconv_silu(const float* __restrict__ zx, const float* __restrict__ cw,
                              const float* __restrict__ cb, float* __restrict__ out) {
  size_t idx = (size_t)blockIdx.x * blockDim.x + threadIdx.x;
  if (idx >= (size_t)MROWS * CONVD) return;
  const int c = (int)(idx % CONVD);
  const int m = (int)(idx / CONVD);
  const int l = m % SEQL;
  float acc = cb[c];
#pragma unroll
  for (int k = 0; k < 4; ++k) {
    const int ls = l + k - 3;
    if (ls >= 0) acc += zx[(size_t)(m + k - 3) * ZXW + DI + c] * cw[c * 4 + k];
  }
  out[idx] = siluf(acc);
}

// ---------------- selective-state scan: block = (batch, head); 256 threads ----------------
// thread t owns p = t>>2 (head dim), 16 states n = (t&3)*16 .. +15.
// Per-step shared vectors (x, B, C) staged in LDS; 64x64 state in registers.
__global__ void k_ssm_scan(const float* __restrict__ zx, const float* __restrict__ xbc,
                           const float* __restrict__ A_log, const float* __restrict__ dt_bias,
                           const float* __restrict__ Dv, float* __restrict__ y) {
  const int b = blockIdx.x >> 3, hh = blockIdx.x & 7;
  const int t = threadIdx.x;
  const int p = t >> 2, q = t & 3, nb = q * 16;
  const float A  = -expf(A_log[hh]);
  const float Dh = Dv[hh];
  const float db = dt_bias[hh];
  __shared__ float sx[HD], sB[DSTATE], sC[DSTATE];
  float h[16];
#pragma unroll
  for (int i = 0; i < 16; ++i) h[i] = 0.f;
  for (int l = 0; l < SEQL; ++l) {
    const size_t row = (size_t)b * SEQL + l;
    const float* xr = xbc + row * CONVD;
    if (t < HD)                       sx[t]            = xr[hh * HD + t];
    else if (t < HD + DSTATE)         sB[t - HD]       = xr[DI + (t - HD)];
    else if (t < HD + 2 * DSTATE)     sC[t - HD - DSTATE] = xr[DI + DSTATE + (t - HD - DSTATE)];
    __syncthreads();
    const float dtr = zx[row * ZXW + DI + CONVD + hh] + db;
    const float dt  = (dtr > 20.f) ? dtr : log1pf(expf(dtr));
    const float dA  = expf(dt * A);
    const float xp = sx[p];
    const float coef = dt * xp;
    float yacc = 0.f;
#pragma unroll
    for (int j = 0; j < 16; ++j) {
      h[j] = dA * h[j] + coef * sB[nb + j];
      yacc += h[j] * sC[nb + j];
    }
    yacc += __shfl_xor(yacc, 1, 32);
    yacc += __shfl_xor(yacc, 2, 32);
    if (q == 0) y[row * DI + hh * HD + p] = yacc + Dh * xp;
    __syncthreads();
  }
}

// ---------------- gated RMSNorm over D_INNER=512 (reads z from zx cols [0,512)) ----------
__global__ void k_gated_rms512(const float* __restrict__ y, const float* __restrict__ zx,
                               const float* __restrict__ w, bf16* __restrict__ out) {
  const int m = blockIdx.x, t = threadIdx.x;
  const float* yr = y + (size_t)m * DI;
  const float* zr = zx + (size_t)m * ZXW;
  const float v0 = yr[t] * siluf(zr[t]);
  const float v1 = yr[t + 256] * siluf(zr[t + 256]);
  __shared__ float red[256];
  red[t] = v0 * v0 + v1 * v1;
  __syncthreads();
  for (int s = 128; s > 0; s >>= 1) { if (t < s) red[t] += red[t + s]; __syncthreads(); }
  const float r = rsqrtf(red[0] / (float)DI + EPSF);
  out[(size_t)m * DI + t]       = tobf(v0 * r * w[t]);
  out[(size_t)m * DI + t + 256] = tobf(v1 * r * w[t + 256]);
}

// ---------------- RMSNorm over D_MODEL=256 ----------------
__global__ void k_rmsnorm256(const float* __restrict__ x, const float* __restrict__ w,
                             float* __restrict__ outF, bf16* __restrict__ outB) {
  const int m = blockIdx.x, t = threadIdx.x;
  const float v = x[(size_t)m * DM + t];
  __shared__ float red[256];
  red[t] = v * v;
  __syncthreads();
  for (int s = 128; s > 0; s >>= 1) { if (t < s) red[t] += red[t + s]; __syncthreads(); }
  const float o = v * rsqrtf(red[0] / (float)DM + EPSF) * w[t];
  if (outF) outF[(size_t)m * DM + t] = o;
  if (outB) outB[(size_t)m * DM + t] = tobf(o);
}

// ---------------- softmax over 50 real cols of [M,64] (scaled by 1/sqrt(256)) -----------
__global__ void k_softmax50(const float* __restrict__ s, bf16* __restrict__ attn) {
  const int m = blockIdx.x, t = threadIdx.x;   // 64 threads
  __shared__ float red[64];
  const float v = (t < 50) ? s[(size_t)m * 64 + t] * 0.0625f : -1e30f;
  red[t] = v;
  __syncthreads();
  for (int k = 32; k > 0; k >>= 1) { if (t < k) red[t] = fmaxf(red[t], red[t + k]); __syncthreads(); }
  const float mx = red[0];
  __syncthreads();
  const float e = (t < 50) ? expf(v - mx) : 0.f;
  red[t] = e;
  __syncthreads();
  for (int k = 32; k > 0; k >>= 1) { if (t < k) red[t] += red[t + k]; __syncthreads(); }
  attn[(size_t)m * 64 + t] = tobf(e / red[0]);
}

// ---------------- elementwise / packing kernels ----------------
__global__ void k_copy_cvt(const float* s, float* dF, bf16* dB, size_t n) {
  size_t i = (size_t)blockIdx.x * blockDim.x + threadIdx.x;
  if (i >= n) return;
  const float v = s[i];
  if (dF) dF[i] = v;
  if (dB) dB[i] = tobf(v);
}
__global__ void k_add(const float* a, const float* b, float* dF, bf16* dB, size_t n) {
  size_t i = (size_t)blockIdx.x * blockDim.x + threadIdx.x;
  if (i >= n) return;
  const float v = a[i] + b[i];
  if (dF) dF[i] = v;
  if (dB) dB[i] = tobf(v);
}
__global__ void k_gate(const float* a, const float* b, const float* g, bf16* dB, size_t n) {
  size_t i = (size_t)blockIdx.x * blockDim.x + threadIdx.x;
  if (i >= n) return;
  dB[i] = tobf((a[i] + b[i]) * g[i]);
}
__global__ void k_scale_shift(const float* x, const float* sc, const float* sh,
                              float* d, size_t n) {
  size_t i = (size_t)blockIdx.x * blockDim.x + threadIdx.x;
  if (i >= n) return;
  d[i] = x[i] * sc[i] + sh[i];
}
__global__ void k_reverse(const float* s, float* dF, bf16* dB) {
  size_t i = (size_t)blockIdx.x * blockDim.x + threadIdx.x;
  if (i >= (size_t)MROWS * DM) return;
  const int d = (int)(i % DM);
  const int m = (int)(i / DM);
  const int b = m / SEQL, l = m % SEQL;
  const float v = s[((size_t)b * SEQL + (SEQL - 1 - l)) * DM + d];
  if (dF) dF[i] = v;
  if (dB) dB[i] = tobf(v);
}
__global__ void k_concat(const float* a, const float* b, bf16* d) {
  size_t i = (size_t)blockIdx.x * blockDim.x + threadIdx.x;
  if (i >= (size_t)MROWS * 512) return;
  const int c = (int)(i % 512);
  const size_t m = i / 512;
  d[i] = tobf(c < 256 ? a[m * 256 + c] : b[m * 256 + (c - 256)]);
}
__global__ void k_cvt_pad_w(const float* s, bf16* d, int N, int Npad, int K) {
  size_t i = (size_t)blockIdx.x * blockDim.x + threadIdx.x;
  if (i >= (size_t)Npad * K) return;
  const int n = (int)(i / K);
  const int k = (int)(i % K);
  d[i] = (n < N) ? tobf(s[(size_t)n * K + k]) : bf0();
}
__global__ void k_pack_conv3(const float* s, bf16* d) {   // [256,256,3] -> [256, 3*256]
  size_t i = (size_t)blockIdx.x * blockDim.x + threadIdx.x;
  if (i >= (size_t)DM * 768) return;
  const int o = (int)(i / 768);
  const int kk = (int)(i % 768);
  const int t = kk / DM, c = kk % DM;
  d[i] = tobf(s[((size_t)o * DM + c) * 3 + t]);
}
__global__ void k_im2col3(const float* x, bf16* d) {      // [B,L,DM] -> [M, 3*DM]
  size_t i = (size_t)blockIdx.x * blockDim.x + threadIdx.x;
  if (i >= (size_t)MROWS * 768) return;
  const int kk = (int)(i % 768);
  const int m = (int)(i / 768);
  const int t = kk / DM, c = kk % DM;
  const int l = m % SEQL, ls = l + t - 1;
  d[i] = (ls >= 0 && ls < SEQL) ? tobf(x[(size_t)(m + t - 1) * DM + c]) : bf0();
}
__global__ void k_pack_confA(const float* cf, bf16* d) {  // [50,256] -> [64,256] pad rows
  size_t i = (size_t)blockIdx.x * blockDim.x + threadIdx.x;
  if (i >= (size_t)64 * DM) return;
  const int k = (int)(i / DM), c = (int)(i % DM);
  d[i] = (k < 50) ? tobf(cf[(size_t)k * DM + c]) : bf0();
}
__global__ void k_pack_confT(const float* cf, bf16* d) {  // [50,256] -> [256,64] transpose+pad
  size_t i = (size_t)blockIdx.x * blockDim.x + threadIdx.x;
  if (i >= (size_t)DM * 64) return;
  const int c = (int)(i / 64), k = (int)(i % 64);
  d[i] = (k < 50) ? tobf(cf[(size_t)k * DM + c]) : bf0();
}

// ---------------- host-side param structs ----------------
struct LinP   { const float *b, *w; };
struct MambaP { const float *A_log, *D, *conv_b, *conv_w, *dt_bias, *in_w, *norm_w, *out_w; };
struct BimambaP {
  const float *bconv_b, *bconv_w; LinP blin; MambaP bm;
  const float *fconv_b, *fconv_w; LinP flin; MambaP fm;
  LinP gate; const float* norm_w; LinP out;
};
struct CondP { LinP kproj; MambaP m; LinP mscale, mshift; const float* norm_w; LinP qproj; };
struct MambaW { bf16 *inw, *outw; };

extern "C" void kernel_launch(void* const* d_in, const int* in_sizes, int n_in,
                              void* d_out, int out_size, void* d_ws, size_t ws_size,
                              hipStream_t stream) {
  (void)in_sizes; (void)n_in; (void)out_size; (void)ws_size;

  // ---- unpack params (JAX pytree order: dicts flattened with sorted keys) ----
  int ip = 0;
  auto F = [&]() { return (const float*)d_in[ip++]; };
  auto rdMamba = [&](MambaP& m) {
    m.A_log = F(); m.D = F(); m.conv_b = F(); m.conv_w = F();
    m.dt_bias = F(); m.in_w = F(); m.norm_w = F(); m.out_w = F();
  };
  const float* x = F();
  const float* confounder = F();
  CondP cs[2];
  for (int i = 0; i < 2; ++i) {
    cs[i].kproj.b = F(); cs[i].kproj.w = F();
    rdMamba(cs[i].m);
    cs[i].mscale.b = F(); cs[i].mscale.w = F();
    cs[i].mshift.b = F(); cs[i].mshift.w = F();
    cs[i].norm_w = F();
    cs[i].qproj.b = F(); cs[i].qproj.w = F();
  }
  LinP f1; f1.b = F(); f1.w = F();
  LinP f2; f2.b = F(); f2.w = F();
  LinP f3; f3.b = F(); f3.w = F();
  BimambaP is[2];
  for (int i = 0; i < 2; ++i) {
    is[i].bconv_b = F(); is[i].bconv_w = F();
    is[i].blin.b = F(); is[i].blin.w = F();
    rdMamba(is[i].bm);
    is[i].fconv_b = F(); is[i].fconv_w = F();
    is[i].flin.b = F(); is[i].flin.w = F();
    rdMamba(is[i].fm);
    is[i].gate.b = F(); is[i].gate.w = F();
    is[i].norm_w = F();
    is[i].out.b = F(); is[i].out.w = F();
  }

  // ---- workspace arena ----
  char* base = (char*)d_ws;
  size_t off = 0;
  auto alloc = [&](size_t bytes) { off = (off + 255) & ~(size_t)255; void* p = base + off; off += bytes; return p; };
  auto allocF = [&](size_t n) { return (float*)alloc(n * 4); };
  auto allocB = [&](size_t n) { return (bf16*)alloc(n * 2); };

  // converted weights
  MambaW mw_csm[2], mw_fm[2], mw_bm[2];
  bf16 *w_gate[2], *w_flin[2], *w_blin[2], *w_out[2], *w_fconv[2], *w_bconv[2];
  bf16 *w_kproj[2], *w_qproj[2], *w_mscale[2], *w_mshift[2];
  for (int i = 0; i < 2; ++i) {
    mw_csm[i].inw = allocB((size_t)ZXW * DM); mw_csm[i].outw = allocB((size_t)DM * DI);
    mw_fm[i].inw  = allocB((size_t)ZXW * DM); mw_fm[i].outw  = allocB((size_t)DM * DI);
    mw_bm[i].inw  = allocB((size_t)ZXW * DM); mw_bm[i].outw  = allocB((size_t)DM * DI);
    w_gate[i] = allocB((size_t)DM * DM); w_flin[i] = allocB((size_t)DM * DM);
    w_blin[i] = allocB((size_t)DM * DM); w_out[i]  = allocB((size_t)DM * DM);
    w_fconv[i] = allocB((size_t)DM * 768); w_bconv[i] = allocB((size_t)DM * 768);
    w_kproj[i] = allocB((size_t)DM * DM); w_qproj[i] = allocB((size_t)DM * DM);
    w_mscale[i] = allocB((size_t)DM * DM); w_mshift[i] = allocB((size_t)DM * DM);
  }
  bf16* w_f1 = allocB((size_t)1024 * 512);
  bf16* w_f2 = allocB((size_t)512 * 1024);
  bf16* w_f3 = allocB((size_t)256 * 512);
  bf16* confA = allocB((size_t)64 * DM);
  bf16* confT = allocB((size_t)DM * 64);
  bf16* dkbf  = allocB((size_t)64 * DM);

  // activations
  float* F_IS = allocF((size_t)MROWS * DM);  bf16* B_IS = allocB((size_t)MROWS * DM);
  float* F_CS = allocF((size_t)MROWS * DM);  bf16* B_CS = allocB((size_t)MROWS * DM);
  float* MS_ZX  = allocF((size_t)MROWS * ZXW);
  float* MS_XBC = allocF((size_t)MROWS * CONVD);
  float* MS_Y   = allocF((size_t)MROWS * DI);
  bf16*  MS_YBF = allocB((size_t)MROWS * DI);
  float* T0 = allocF((size_t)MROWS * DM); float* T1 = allocF((size_t)MROWS * DM);
  float* T2 = allocF((size_t)MROWS * DM); float* T3 = allocF((size_t)MROWS * DM);
  bf16* TB0 = allocB((size_t)MROWS * DM); bf16* TB1 = allocB((size_t)MROWS * DM);
  bf16* TB2 = allocB((size_t)MROWS * DM);
  bf16* A768   = allocB((size_t)MROWS * 768);
  float* SCORES = allocF((size_t)MROWS * 64);
  bf16*  ATTN   = allocB((size_t)MROWS * 64);
  bf16* H512  = allocB((size_t)MROWS * 512);
  bf16* H1024 = allocB((size_t)MROWS * 1024);

  // ---- launch helpers ----
  auto cdiv = [](size_t a, size_t b) { return (unsigned)((a + b - 1) / b); };
  auto gemm = [&](const bf16* A, const bf16* W, const float* bias, float* Cf, bf16* Cb,
                  int M, int N, int K, int act) {
    const int strips = (M / 16) * (N / 64);
    wmma_gemm_kernel<<<dim3((strips + 7) / 8), 256, 0, stream>>>(A, W, bias, Cf, Cb, M, N, K, act);
  };
  auto cvtw = [&](const float* s, bf16* d, int N, int Npad, int K) {
    k_cvt_pad_w<<<cdiv((size_t)Npad * K, 256), 256, 0, stream>>>(s, d, N, Npad, K);
  };

  // ---- convert / pack all weights ----
  for (int i = 0; i < 2; ++i) {
    cvtw(cs[i].m.in_w, mw_csm[i].inw, ZX_REAL, ZXW, DM);
    cvtw(cs[i].m.out_w, mw_csm[i].outw, DM, DM, DI);
    cvtw(is[i].fm.in_w, mw_fm[i].inw, ZX_REAL, ZXW, DM);
    cvtw(is[i].fm.out_w, mw_fm[i].outw, DM, DM, DI);
    cvtw(is[i].bm.in_w, mw_bm[i].inw, ZX_REAL, ZXW, DM);
    cvtw(is[i].bm.out_w, mw_bm[i].outw, DM, DM, DI);
    cvtw(is[i].gate.w, w_gate[i], DM, DM, DM);
    cvtw(is[i].flin.w, w_flin[i], DM, DM, DM);
    cvtw(is[i].blin.w, w_blin[i], DM, DM, DM);
    cvtw(is[i].out.w,  w_out[i],  DM, DM, DM);
    k_pack_conv3<<<cdiv((size_t)DM * 768, 256), 256, 0, stream>>>(is[i].fconv_w, w_fconv[i]);
    k_pack_conv3<<<cdiv((size_t)DM * 768, 256), 256, 0, stream>>>(is[i].bconv_w, w_bconv[i]);
    cvtw(cs[i].kproj.w,  w_kproj[i],  DM, DM, DM);
    cvtw(cs[i].qproj.w,  w_qproj[i],  DM, DM, DM);
    cvtw(cs[i].mscale.w, w_mscale[i], DM, DM, DM);
    cvtw(cs[i].mshift.w, w_mshift[i], DM, DM, DM);
  }
  cvtw(f1.w, w_f1, 1024, 1024, 512);
  cvtw(f2.w, w_f2, 512, 512, 1024);
  cvtw(f3.w, w_f3, 256, 256, 512);
  k_pack_confA<<<cdiv((size_t)64 * DM, 256), 256, 0, stream>>>(confounder, confA);
  k_pack_confT<<<cdiv((size_t)DM * 64, 256), 256, 0, stream>>>(confounder, confT);

  // ---- mamba2: u_bf [M,256] bf16 -> outF [M,256] f32 ----
  auto run_mamba2 = [&](const bf16* u_bf, const MambaW& w, const MambaP& p, float* outF) {
    gemm(u_bf, w.inw, nullptr, MS_ZX, nullptr, MROWS, ZXW, DM, 0);          // zxbcdt
    k_dwconv_silu<<<cdiv((size_t)MROWS * CONVD, 256), 256, 0, stream>>>(MS_ZX, p.conv_w, p.conv_b, MS_XBC);
    k_ssm_scan<<<BSZ * NH, 256, 0, stream>>>(MS_ZX, MS_XBC, p.A_log, p.dt_bias, p.D, MS_Y);
    k_gated_rms512<<<MROWS, 256, 0, stream>>>(MS_Y, MS_ZX, p.norm_w, MS_YBF);
    gemm(MS_YBF, w.outw, nullptr, outF, nullptr, MROWS, DM, DI, 0);        // out proj
  };

  const size_t NMD = (size_t)MROWS * DM;
  // ---- init streams: is_out = cs_out = x ----
  k_copy_cvt<<<cdiv(NMD, 256), 256, 0, stream>>>(x, F_IS, B_IS, NMD);
  k_copy_cvt<<<cdiv(NMD, 256), 256, 0, stream>>>(x, F_CS, B_CS, NMD);

  // ---- bimamba blocks (is stream) ----
  for (int i = 0; i < 2; ++i) {
    gemm(B_IS, w_gate[i], is[i].gate.b, T0, nullptr, MROWS, DM, DM, 1);    // gate=silu(..)
    k_im2col3<<<cdiv((size_t)MROWS * 768, 256), 256, 0, stream>>>(F_IS, A768);
    gemm(A768, w_fconv[i], is[i].fconv_b, T1, nullptr, MROWS, DM, 768, 0); // fconv
    k_add<<<cdiv(NMD, 256), 256, 0, stream>>>(F_IS, T1, T2, TB0, NMD);     // xf
    gemm(TB0, w_flin[i], is[i].flin.b, nullptr, TB1, MROWS, DM, DM, 0);    // flin
    run_mamba2(TB1, mw_fm[i], is[i].fm, T1);                               // of -> T1
    k_reverse<<<cdiv(NMD, 256), 256, 0, stream>>>(F_IS, T2, nullptr);      // xr -> T2
    k_im2col3<<<cdiv((size_t)MROWS * 768, 256), 256, 0, stream>>>(T2, A768);
    gemm(A768, w_bconv[i], is[i].bconv_b, T3, nullptr, MROWS, DM, 768, 0); // bconv
    k_add<<<cdiv(NMD, 256), 256, 0, stream>>>(T2, T3, nullptr, TB0, NMD);  // xb (bf only)
    gemm(TB0, w_blin[i], is[i].blin.b, nullptr, TB1, MROWS, DM, DM, 0);    // blin
    run_mamba2(TB1, mw_bm[i], is[i].bm, T2);                               // ob_rev -> T2
    k_reverse<<<cdiv(NMD, 256), 256, 0, stream>>>(T2, T3, nullptr);        // ob -> T3
    k_gate<<<cdiv(NMD, 256), 256, 0, stream>>>(T1, T3, T0, TB0, NMD);      // (of+ob)*gate
    gemm(TB0, w_out[i], is[i].out.b, T2, nullptr, MROWS, DM, DM, 0);       // out proj
    k_rmsnorm256<<<MROWS, 256, 0, stream>>>(T2, is[i].norm_w, T3, nullptr);
    k_add<<<cdiv(NMD, 256), 256, 0, stream>>>(F_IS, T3, F_IS, B_IS, NMD);  // residual
  }

  // ---- conditioning blocks (cs stream) ----
  for (int i = 0; i < 2; ++i) {
    gemm(confA, w_kproj[i], cs[i].kproj.b, nullptr, dkbf, 64, DM, DM, 0);  // dk [64,256]
    gemm(B_CS, w_qproj[i], cs[i].qproj.b, nullptr, TB0, MROWS, DM, DM, 0); // q
    gemm(TB0, dkbf, nullptr, SCORES, nullptr, MROWS, 64, DM, 0);           // q @ dk^T
    k_softmax50<<<MROWS, 64, 0, stream>>>(SCORES, ATTN);
    gemm(ATTN, confT, nullptr, nullptr, TB1, MROWS, DM, 64, 0);            // c = attn@conf
    gemm(TB1, w_mscale[i], cs[i].mscale.b, T0, nullptr, MROWS, DM, DM, 3); // sigmoid scale
    gemm(TB1, w_mshift[i], cs[i].mshift.b, T1, nullptr, MROWS, DM, DM, 0); // shift
    k_scale_shift<<<cdiv(NMD, 256), 256, 0, stream>>>(F_CS, T0, T1, T2, NMD);
    k_rmsnorm256<<<MROWS, 256, 0, stream>>>(T2, cs[i].norm_w, nullptr, TB2);
    run_mamba2(TB2, mw_csm[i], cs[i].m, T0);                               // mamba out
    k_add<<<cdiv(NMD, 256), 256, 0, stream>>>(F_CS, T0, F_CS, B_CS, NMD);  // residual
  }

  // ---- head MLP ----
  k_concat<<<cdiv((size_t)MROWS * 512, 256), 256, 0, stream>>>(F_IS, F_CS, H512);
  gemm(H512,  w_f1, f1.b, nullptr, H1024, MROWS, 1024, 512, 2);            // tanh
  gemm(H1024, w_f2, f2.b, nullptr, H512,  MROWS, 512, 1024, 2);            // tanh
  gemm(H512,  w_f3, f3.b, (float*)d_out, nullptr, MROWS, 256, 512, 0);     // final
}
